// LinearAttention_28381143891997
// MI455X (gfx1250) — matrix operand order repaired and checked by reading, not compiled, not run on hardware
//
#include <hip/hip_runtime.h>
#include <hip/hip_bf16.h>

// ---------------------------------------------------------------------------
// MI455X (gfx1250): all matmuls via v_wmma_f32_16x16x32_f16, wave32,
// fragments in contiguous k = 16*half + s order; attention K/V staged by the Tensor Data
// Mover (tensor_load_to_lds + s_wait_tensorcnt), double-buffered.
// ---------------------------------------------------------------------------

typedef __attribute__((ext_vector_type(16))) _Float16 v16h;
typedef __attribute__((ext_vector_type(8)))  float    v8f;
typedef unsigned int u32x4 __attribute__((ext_vector_type(4)));
typedef int          i32x8 __attribute__((ext_vector_type(8)));
typedef int          i32x4 __attribute__((ext_vector_type(4)));

#define DIM      1024
#define SEQ      2048
#define BATCH    2
#define HEADS    16
#define DHEAD    64
#define MTOT     (BATCH * SEQ)        // 4096 rows
#define N_QKV    (3 * DIM)            // 3072
#define ATTN_SCALE 0.125f             // 64^-0.5

#if __has_builtin(__builtin_amdgcn_tensor_load_to_lds)
#define HAVE_TDM 1
#else
#define HAVE_TDM 0                    // host parse / older toolchain fallback
#endif

// ---------------- workspace layout (bytes) ----------------
#define XB_OFF   ((size_t)0)                          // x as f16      [4096,1024]
#define WQ_OFF   ((size_t)8388608)                    // Wqkv^T f16    [3072,1024]
#define WO_OFF   ((size_t)14680064)                   // Wout^T f16    [1024,1024]
#define QKV_OFF  ((size_t)16777216)                   // qkv f16       [4096,3072]
#define AO_OFF   ((size_t)41943040)                   // attn out f32  [4096,1024]
#define NM_OFF   ((size_t)58720256)                   // normed f16    [4096,1024]

// ---------------------------------------------------------------------------
// WMMA helper + fragment loaders (contiguous k = 16*half + s order in A and B)
// ---------------------------------------------------------------------------
__device__ __forceinline__ v8f wmma_ff(v16h a, v16h b, v8f c) {
  return __builtin_amdgcn_wmma_f32_16x16x32_f16(false, a, false, b,
                                                (short)0, c, false, false);
}

// A fragment: 16x32 f16, row-major tile (ld elems/row).
__device__ __forceinline__ v16h frag_A(const _Float16* p, int row0, int k0, int ld) {
  const int lane = threadIdx.x & 31;
  const int r = lane & 15, hh = lane >> 4;
  v16h a;
#pragma unroll
  for (int i = 0; i < 8; ++i) {
    const int kb = hh * 16 + i * 2;
    const _Float16* q = p + (row0 + r) * ld + k0 + kb;
    a[2 * i]     = q[0];
    a[2 * i + 1] = q[1];
  }
  return a;
}

// B fragment from a TRANSPOSED tile (holds B^T as [N][K] rows) -> paired reads.
__device__ __forceinline__ v16h frag_BT(const _Float16* p, int n0, int k0, int ld) {
  const int lane = threadIdx.x & 31;
  const int r = lane & 15, hh = lane >> 4;
  v16h b;
#pragma unroll
  for (int i = 0; i < 8; ++i) {
    const _Float16* q = p + (n0 + r) * ld + k0 + hh * 16 + i * 2;
    b[2 * i]     = q[0];
    b[2 * i + 1] = q[1];
  }
  return b;
}

// B fragment from a row-major [K][N] tile (V in attention).
__device__ __forceinline__ v16h frag_Bn(const _Float16* p, int k0, int n0, int ld) {
  const int lane = threadIdx.x & 31;
  const int r = lane & 15, hh = lane >> 4;
  v16h b;
#pragma unroll
  for (int i = 0; i < 8; ++i) {
    const int kk = k0 + hh * 16 + i * 2;
    b[2 * i]     = p[kk * ld + n0 + r];
    b[2 * i + 1] = p[(kk + 1) * ld + n0 + r];
  }
  return b;
}

// ---------------------------------------------------------------------------
// TDM: issue one 2D tile load (rows x cols f16, row_stride in elements)
// into LDS. Descriptor per CDNA5 ISA.
// ---------------------------------------------------------------------------
#if HAVE_TDM
__device__ __forceinline__ void tdm_load_2d(const _Float16* gsrc,
                                            _Float16* ldst,
                                            int rows, int cols, int row_stride) {
  unsigned long long ga = (unsigned long long)(size_t)gsrc;
  u32x4 g0;
  g0[0] = 1u;                                        // count=1, user mode
  g0[1] = (unsigned int)(size_t)ldst;                // lds_addr  [63:32]
  g0[2] = (unsigned int)(ga & 0xffffffffu);          // global_addr [95:64]
  g0[3] = (unsigned int)((ga >> 32) & 0x01ffffffu)   // global_addr [120:96]
        | (2u << 30);                                // type=2 ("image")
  i32x8 g1;
  g1[0] = 1 << 16;          // wg_mask=0, data_size=1 (2 bytes)
  g1[1] = cols << 16;       // tensor_dim0        (bits 79:48)
  g1[2] = rows << 16;       // tensor_dim1        (bits 111:80)
  g1[3] = cols << 16;       // tile_dim0          (bits 127:112)
  g1[4] = rows;             // tile_dim1          (bits 143:128)
  g1[5] = row_stride;       // tensor_dim0_stride (bits 207:160, low 32)
  g1[6] = 0;
  g1[7] = 0;
  i32x4 z4 = {0, 0, 0, 0};
  i32x8 z8 = {0, 0, 0, 0, 0, 0, 0, 0};
  __builtin_amdgcn_tensor_load_to_lds(g0, g1, z4, z4, z8, 0);
}
#endif

// ---------------------------------------------------------------------------
// One-time precision conversion kernels
// ---------------------------------------------------------------------------
__global__ void cvt_f16_kernel(const float* __restrict__ in,
                               _Float16* __restrict__ out, int n) {
  int i = (blockIdx.x * blockDim.x + threadIdx.x) * 4;
  if (i < n) {
    float4 v = *(const float4*)(in + i);
    out[i + 0] = (_Float16)v.x;
    out[i + 1] = (_Float16)v.y;
    out[i + 2] = (_Float16)v.z;
    out[i + 3] = (_Float16)v.w;
  }
}

__global__ void cvt_transpose_kernel(const float* __restrict__ in,
                                     _Float16* __restrict__ out,
                                     int rows, int cols) {
  int idx = blockIdx.x * blockDim.x + threadIdx.x;
  if (idx < rows * cols) {
    int k = idx / cols, n = idx % cols;
    out[(size_t)n * rows + k] = (_Float16)in[idx];
  }
}

// ---------------------------------------------------------------------------
// Tiled WMMA GEMM: C[M][N] = A[M][1024] * B[1024][N]  (B supplied transposed)
// block = 256 threads (8 waves, 4x2), block tile 128x128, K-step 64,
// wave tile 32x64, double-buffered register-staged LDS pipeline.
// ---------------------------------------------------------------------------
template <bool F16OUT>
__global__ void wmma_gemm_kernel(const _Float16* __restrict__ A,
                                 const _Float16* __restrict__ BT,
                                 void* __restrict__ C,
                                 const float* __restrict__ bias,
                                 int ldc) {
  __shared__ __align__(16) _Float16 As[2][128 * 64];   // 2 x 16 KB
  __shared__ __align__(16) _Float16 Bs[2][128 * 64];   // 2 x 16 KB (B^T tiles)

  const int n0 = blockIdx.x * 128;
  const int m0 = blockIdx.y * 128;
  const int t  = threadIdx.x;
  const int w  = t >> 5;
  const int waveM = w >> 1;      // 0..3 -> 32-row slab
  const int waveN = w & 1;       // 0..1 -> 64-col slab
  const int lane = t & 31;
  const int r = lane & 15, hh = lane >> 4;

  // staging: each thread owns 32 A elems + 32 B elems (4 uint4 each)
  const int srow = t >> 1;             // 0..127
  const int scol = (t & 1) * 32;       // 0 / 32
  uint4 ra[4], rb[4];

  auto load_tiles = [&](int k0) {
#pragma unroll
    for (int i = 0; i < 4; ++i) {
      ra[i] = *(const uint4*)(A  + (size_t)(m0 + srow) * 1024 + k0 + scol + i * 8);
      rb[i] = *(const uint4*)(BT + (size_t)(n0 + srow) * 1024 + k0 + scol + i * 8);
    }
  };
  auto store_tiles = [&](int buf) {
#pragma unroll
    for (int i = 0; i < 4; ++i) {
      *(uint4*)(&As[buf][srow * 64 + scol + i * 8]) = ra[i];
      *(uint4*)(&Bs[buf][srow * 64 + scol + i * 8]) = rb[i];
    }
  };

  v8f acc[2][4] = {};

  load_tiles(0);
  store_tiles(0);

  for (int kt = 0; kt < 16; ++kt) {
    __syncthreads();
    if (kt + 1 < 16) load_tiles((kt + 1) * 64);   // in flight during compute
    const int buf = kt & 1;
#pragma unroll
    for (int kk = 0; kk < 64; kk += 32) {
      v16h a0 = frag_A(As[buf], waveM * 32,      kk, 64);
      v16h a1 = frag_A(As[buf], waveM * 32 + 16, kk, 64);
#pragma unroll
      for (int ni = 0; ni < 4; ++ni) {
        v16h b = frag_BT(Bs[buf], waveN * 64 + ni * 16, kk, 64);
        acc[0][ni] = wmma_ff(a0, b, acc[0][ni]);
        acc[1][ni] = wmma_ff(a1, b, acc[1][ni]);
      }
    }
    if (kt + 1 < 16) store_tiles((kt + 1) & 1);
  }

  // Epilogue: C layout m = half*8 + vgpr, n = lane&15
#pragma unroll
  for (int mi = 0; mi < 2; ++mi)
#pragma unroll
    for (int ni = 0; ni < 4; ++ni)
#pragma unroll
      for (int g = 0; g < 8; ++g) {
        int row = m0 + waveM * 32 + mi * 16 + hh * 8 + g;
        int col = n0 + waveN * 64 + ni * 16 + r;
        if (F16OUT) {
          ((_Float16*)C)[(size_t)row * ldc + col] = (_Float16)acc[mi][ni][g];
        } else {
          ((float*)C)[(size_t)row * ldc + col] = acc[mi][ni][g] + bias[col];
        }
      }
}

// ---------------------------------------------------------------------------
// Fused relu-attention: out[i,:] = sum_j relu(scale * q_i.k_j) * v_j
// block = 128 threads (4 waves) per (b, head, 64-row i-tile); Q resident.
// K/V tiles double-buffered via the Tensor Data Mover: wave 0 issues two
// tensor_load_to_lds per j-tile, s_wait_tensorcnt(2) keeps the next pair in
// flight while the current buffer is consumed. Barrier at loop top ensures
// no wave still reads the buffer the next TDM pair will overwrite.
// ---------------------------------------------------------------------------
__global__ void attn_kernel(const _Float16* __restrict__ qkv,
                            float* __restrict__ out) {
  const int i0   = blockIdx.x * 64;
  const int bh   = blockIdx.y;
  const int b    = bh >> 4, head = bh & 15;
  const int tok0 = b * SEQ;
  const int qc = head * DHEAD, kc = DIM + head * DHEAD, vc = 2 * DIM + head * DHEAD;

  __shared__ __align__(16) _Float16 Qs[64 * 64];        // 8 KB
  __shared__ __align__(16) _Float16 Ks[2][64 * 64];     // 16 KB (j x d == B^T layout)
  __shared__ __align__(16) _Float16 Vs[2][64 * 64];     // 16 KB (j x d row-major)
  __shared__ __align__(16) _Float16 Ss[4][16 * 64];     // 8 KB per-wave scores

  const int t = threadIdx.x;       // 0..127
  const int w = t >> 5;            // wave -> Q rows [16w,16w+16)
  const int lane = t & 31;
  const int r = lane & 15, hh = lane >> 4;

  // stage Q tile once (vector loads)
#pragma unroll
  for (int ii = 0; ii < 4; ++ii) {
    int idx = t + 128 * ii, row = idx >> 3, cg = (idx & 7) * 8;
    *(uint4*)(Qs + row * 64 + cg) =
        *(const uint4*)(qkv + (size_t)(tok0 + i0 + row) * N_QKV + qc + cg);
  }

#if HAVE_TDM
  if (w == 0) {   // uniform branch: only wave 0 issues TDM descriptors
    const _Float16* base0 = qkv + (size_t)tok0 * N_QKV;
    tdm_load_2d(base0 + kc, &Ks[0][0], 64, 64, N_QKV);
    tdm_load_2d(base0 + vc, &Vs[0][0], 64, 64, N_QKV);
  }
#else
  uint4 rk[4], rv[4];
  auto load_kv = [&](int j0) {
#pragma unroll
    for (int ii = 0; ii < 4; ++ii) {
      int idx = t + 128 * ii, row = idx >> 3, cg = (idx & 7) * 8;
      size_t base = (size_t)(tok0 + j0 + row) * N_QKV;
      rk[ii] = *(const uint4*)(qkv + base + kc + cg);
      rv[ii] = *(const uint4*)(qkv + base + vc + cg);
    }
  };
  auto store_kv = [&](int buf) {
#pragma unroll
    for (int ii = 0; ii < 4; ++ii) {
      int idx = t + 128 * ii, row = idx >> 3, cg = (idx & 7) * 8;
      *(uint4*)(&Ks[buf][row * 64 + cg]) = rk[ii];
      *(uint4*)(&Vs[buf][row * 64 + cg]) = rv[ii];
    }
  };
  load_kv(0);
  store_kv(0);
#endif

  v8f o[4] = {};

  for (int jt = 0; jt < 32; ++jt) {
    __syncthreads();   // all waves done with buffer (jt+1)&1 from iter jt-1
#if HAVE_TDM
    if (w == 0) {
      if (jt + 1 < 32) {
        const _Float16* nb = qkv + (size_t)(tok0 + (jt + 1) * 64) * N_QKV;
        tdm_load_2d(nb + kc, &Ks[(jt + 1) & 1][0], 64, 64, N_QKV);
        tdm_load_2d(nb + vc, &Vs[(jt + 1) & 1][0], 64, 64, N_QKV);
        __builtin_amdgcn_s_wait_tensorcnt(2);   // current pair complete
      } else {
        __builtin_amdgcn_s_wait_tensorcnt(0);
      }
    }
    __syncthreads();   // publish current buffer to all waves
#else
    if (jt + 1 < 32) load_kv((jt + 1) * 64);
#endif
    const int buf = jt & 1;

    // S = Q(16x64) @ K^T ; Ks rows are j -> already B^T layout
    v8f s[4] = {};
#pragma unroll
    for (int kk = 0; kk < 64; kk += 32) {
      v16h a = frag_A(Qs, w * 16, kk, 64);
#pragma unroll
      for (int nf = 0; nf < 4; ++nf) {
        v16h bb = frag_BT(Ks[buf], nf * 16, kk, 64);
        s[nf] = wmma_ff(a, bb, s[nf]);
      }
    }

    // relu + scale, spill to per-wave LDS (same-wave LDS RAW is in-order)
#pragma unroll
    for (int nf = 0; nf < 4; ++nf)
#pragma unroll
      for (int g = 0; g < 8; ++g) {
        float v = s[nf][g];
        v = v > 0.0f ? v * ATTN_SCALE : 0.0f;
        Ss[w][(hh * 8 + g) * 64 + nf * 16 + r] = (_Float16)v;
      }

    // O += S(16x64) @ V(64x64)
#pragma unroll
    for (int kk = 0; kk < 64; kk += 32) {
      v16h a2 = frag_A(Ss[w], 0, kk, 64);
#pragma unroll
      for (int df = 0; df < 4; ++df) {
        v16h bb = frag_Bn(Vs[buf], kk, df * 16, 64);
        o[df] = wmma_ff(a2, bb, o[df]);
      }
    }

#if !HAVE_TDM
    if (jt + 1 < 32) store_kv((jt + 1) & 1);
#endif
  }

  // merged-head f32 output (pre-LayerNorm)
#pragma unroll
  for (int df = 0; df < 4; ++df)
#pragma unroll
    for (int g = 0; g < 8; ++g) {
      int row = tok0 + i0 + w * 16 + hh * 8 + g;
      int col = head * DHEAD + df * 16 + r;
      out[(size_t)row * DIM + col] = o[df][g];
    }
}

// ---------------------------------------------------------------------------
// LayerNorm over 1024 elems, one 256-thread block per row, f16 output.
// ---------------------------------------------------------------------------
__global__ void ln_kernel(const float* __restrict__ x,
                          const float* __restrict__ gamma,
                          const float* __restrict__ beta,
                          _Float16* __restrict__ y) {
  __shared__ float rs[256], rs2[256];
  const int row = blockIdx.x;
  const int t = threadIdx.x;
  const float4 v = *(const float4*)(x + (size_t)row * DIM + t * 4);
  rs[t]  = v.x + v.y + v.z + v.w;
  rs2[t] = v.x * v.x + v.y * v.y + v.z * v.z + v.w * v.w;
  __syncthreads();
  for (int off = 128; off > 0; off >>= 1) {
    if (t < off) { rs[t] += rs[t + off]; rs2[t] += rs2[t + off]; }
    __syncthreads();
  }
  const float mu   = rs[0] * (1.0f / DIM);
  const float var  = rs2[0] * (1.0f / DIM) - mu * mu;
  const float rstd = rsqrtf(var + 1e-5f);
  const float vv[4] = {v.x, v.y, v.z, v.w};
#pragma unroll
  for (int e = 0; e < 4; ++e) {
    int c = t * 4 + e;
    y[(size_t)row * DIM + c] = (_Float16)((vv[e] - mu) * rstd * gamma[c] + beta[c]);
  }
}

// ---------------------------------------------------------------------------
extern "C" void kernel_launch(void* const* d_in, const int* in_sizes, int n_in,
                              void* d_out, int out_size, void* d_ws, size_t ws_size,
                              hipStream_t stream) {
  (void)in_sizes; (void)n_in; (void)out_size; (void)ws_size;
  const float* x     = (const float*)d_in[0];
  const float* W_qkv = (const float*)d_in[1];
  const float* gamma = (const float*)d_in[2];
  const float* beta  = (const float*)d_in[3];
  const float* W_out = (const float*)d_in[4];
  const float* b_out = (const float*)d_in[5];
  float* out = (float*)d_out;

  char* ws = (char*)d_ws;
  _Float16* xb     = (_Float16*)(ws + XB_OFF);
  _Float16* wqkvT  = (_Float16*)(ws + WQ_OFF);
  _Float16* woutT  = (_Float16*)(ws + WO_OFF);
  _Float16* qkv    = (_Float16*)(ws + QKV_OFF);
  float*    attn_o = (float*)   (ws + AO_OFF);
  _Float16* normed = (_Float16*)(ws + NM_OFF);

  // 1) precision conversion (one-time; weights transposed for B^T staging)
  cvt_f16_kernel<<<4096, 256, 0, stream>>>(x, xb, MTOT * DIM);
  cvt_transpose_kernel<<<12288, 256, 0, stream>>>(W_qkv, wqkvT, DIM, N_QKV);
  cvt_transpose_kernel<<<4096, 256, 0, stream>>>(W_out, woutT, DIM, DIM);

  // 2) QKV projection: [4096,1024] @ [1024,3072] -> f16 qkv
  wmma_gemm_kernel<true><<<dim3(N_QKV / 128, MTOT / 128), 256, 0, stream>>>(
      xb, wqkvT, (void*)qkv, nullptr, N_QKV);

  // 3) fused relu-attention (TDM-staged K/V) -> merged-head f32
  attn_kernel<<<dim3(SEQ / 64, BATCH * HEADS), 128, 0, stream>>>(qkv, attn_o);

  // 4) LayerNorm -> f16
  ln_kernel<<<MTOT, 256, 0, stream>>>(attn_o, gamma, beta, normed);

  // 5) output projection + bias: [4096,1024] @ [1024,1024] -> f32 d_out
  wmma_gemm_kernel<false><<<dim3(DIM / 128, MTOT / 128), 256, 0, stream>>>(
      normed, woutT, (void*)out, b_out, DIM);
}
